// MoEGate_25640954757689
// MI455X (gfx1250) — matrix operand order repaired and checked by reading, not compiled, not run on hardware
//
#include <hip/hip_runtime.h>
#include <hip/hip_bf16.h>

typedef __attribute__((ext_vector_type(16))) __bf16 v16bf;
typedef __attribute__((ext_vector_type(8)))  __bf16 v8bf;
typedef __attribute__((ext_vector_type(4)))  __bf16 v4bf;
typedef __attribute__((ext_vector_type(8)))  float  v8f;

#define E_        64
#define K_        10
#define D_        2048
#define NTOK      16384          // B*T
#define BTK       163840         // B*T*K
#define KSLICE    256            // K elements of gate_w staged in LDS per pass
#define WROW      (KSLICE + 8)   // bf16 elems per LDS row (pad -> 264, stride 528B, breaks bank alignment)
#define LROW      68             // fp32 logits row stride (64 + 4 pad dwords)
#define TOK_PER_BLOCK 128
#define TOK_PER_WAVE  16

// Workspace accumulators: [0..63]=sum P_real, [64..127]=counts, [128]=sum lse^2, [129]=null slots
__global__ void moe_zero_acc(float* __restrict__ g) {
    if (threadIdx.x < 130) g[threadIdx.x] = 0.0f;
}

// A fragment (16-bit A 16x32 layout): this lane's 16 contiguous-pair fp32 -> bf16
__device__ __forceinline__ v16bf load_a_frag(const float* __restrict__ ap) {
    float4 a0 = *(const float4*)(ap + 0);
    float4 a1 = *(const float4*)(ap + 4);
    float4 a2 = *(const float4*)(ap + 16);
    float4 a3 = *(const float4*)(ap + 20);
    v16bf a;
    a[0]=(__bf16)a0.x; a[1]=(__bf16)a0.y; a[2]=(__bf16)a0.z;  a[3]=(__bf16)a0.w;
    a[4]=(__bf16)a1.x; a[5]=(__bf16)a1.y; a[6]=(__bf16)a1.z;  a[7]=(__bf16)a1.w;
    a[8]=(__bf16)a2.x; a[9]=(__bf16)a2.y; a[10]=(__bf16)a2.z; a[11]=(__bf16)a2.w;
    a[12]=(__bf16)a3.x;a[13]=(__bf16)a3.y;a[14]=(__bf16)a3.z; a[15]=(__bf16)a3.w;
    return a;
}

// B fragment (16-bit B 32x16 layout) from LDS: 16 bf16 = two 16B loads
__device__ __forceinline__ v16bf load_b_frag(const __bf16* bp) {
    v8bf lo  = *(const v8bf*)bp;
    v8bf hi8 = *(const v8bf*)(bp + 16);
    return __builtin_shufflevector(lo, hi8, 0,1,2,3,4,5,6,7,8,9,10,11,12,13,14,15);
}

__global__ __launch_bounds__(256)
void moe_gate_main(const float* __restrict__ x,
                   const float* __restrict__ gate_w,
                   const float* __restrict__ logit_bias,
                   const float* __restrict__ null_logit,
                   float* __restrict__ out_idx,
                   float* __restrict__ out_w,
                   float* __restrict__ out_isnull,
                   float* __restrict__ g_acc)
{
    // Reused buffer: gate_w bf16 slice (64*264*2 = 33792B) then logits (128*68*4 = 34816B)
    __shared__ __align__(16) char smem_raw[TOK_PER_BLOCK * LROW * 4];
    __shared__ float blk_P[E_];
    __shared__ float blk_cnt[E_];
    __shared__ float blk_misc[2];   // [0]=lse^2 sum, [1]=null-slot count

    const int tid    = threadIdx.x;
    const int wave   = tid >> 5;
    const int lane   = tid & 31;
    const int lane16 = lane & 15;
    const bool hi    = lane >= 16;

    if (tid < E_) { blk_P[tid] = 0.0f; blk_cnt[tid] = 0.0f; }
    if (tid < 2)  blk_misc[tid] = 0.0f;

    const int tok0 = blockIdx.x * TOK_PER_BLOCK;
    const int wtok = tok0 + wave * TOK_PER_WAVE;

    __bf16* wlds       = (__bf16*)smem_raw;
    float*  logits_lds = (float*)smem_raw;

    // Row of x this lane supplies for the A fragment:
    // lanes 0-15: M=lane, K = {0..7, 16..23}; lanes 16-31: M=lane-16, K = {8..15, 24..31}
    const float* xrow = x + (size_t)(wtok + lane16) * D_ + (hi ? 8 : 0);
    const __bf16* bbase = wlds + lane16 * WROW + (hi ? 8 : 0);

    v8f acc0 = {}, acc1 = {}, acc2 = {}, acc3 = {};

    for (int s = 0; s < D_ / KSLICE; ++s) {
        __syncthreads();  // previous slice fully consumed (also covers blk_* init on s==0)
        // Stage gate_w[:, s*256 .. +256) as bf16 into LDS; float4 loads, 8B packed stores
        {
            const float* gbase = gate_w + s * KSLICE;
            #pragma unroll
            for (int it = 0; it < (E_ * KSLICE / 4) / 256; ++it) {   // 16 iters
                int q  = it * 256 + tid;
                int e  = q >> 6;            // 64 float4-chunks per 256-float row
                int c4 = (q & 63) * 4;
                float4 v = *(const float4*)(gbase + (size_t)e * D_ + c4);
                v4bf w;
                w[0] = (__bf16)v.x; w[1] = (__bf16)v.y;
                w[2] = (__bf16)v.z; w[3] = (__bf16)v.w;
                *(v4bf*)(wlds + e * WROW + c4) = w;   // 8B aligned
            }
        }
        __syncthreads();

        // Software-pipelined A: convert kt+1's fragment while WMMAs consume kt's.
        v16bf a_cur = load_a_frag(xrow + s * KSLICE);
        #pragma unroll
        for (int kt = 0; kt < KSLICE / 32; ++kt) {
            v16bf a_next;
            if (kt < KSLICE / 32 - 1)
                a_next = load_a_frag(xrow + s * KSLICE + (kt + 1) * 32);

            // All 4 B fragments up front -> one ds_load clause, staggered waits.
            const int kc = kt * 32;
            v16bf b0 = load_b_frag(bbase + 0 * 16 * WROW + kc);
            v16bf b1 = load_b_frag(bbase + 1 * 16 * WROW + kc);
            v16bf b2 = load_b_frag(bbase + 2 * 16 * WROW + kc);
            v16bf b3 = load_b_frag(bbase + 3 * 16 * WROW + kc);

            acc0 = __builtin_amdgcn_wmma_f32_16x16x32_bf16(false, a_cur, false, b0, (short)0, acc0, false, false);
            acc1 = __builtin_amdgcn_wmma_f32_16x16x32_bf16(false, a_cur, false, b1, (short)0, acc1, false, false);
            acc2 = __builtin_amdgcn_wmma_f32_16x16x32_bf16(false, a_cur, false, b2, (short)0, acc2, false, false);
            acc3 = __builtin_amdgcn_wmma_f32_16x16x32_bf16(false, a_cur, false, b3, (short)0, acc3, false, false);

            if (kt < KSLICE / 32 - 1) a_cur = a_next;
        }
    }

    __syncthreads();   // GEMM done everywhere; safe to overwrite wlds with logits
    {
        // C layout: reg r <-> row (r + 8*hi), col = lane16 within each 16-wide N tile
        float b0 = logit_bias[ 0 + lane16];
        float b1 = logit_bias[16 + lane16];
        float b2 = logit_bias[32 + lane16];
        float b3 = logit_bias[48 + lane16];
        const int mbase = wave * TOK_PER_WAVE + (hi ? 8 : 0);
        #pragma unroll
        for (int r = 0; r < 8; ++r) {
            float* lr = logits_lds + (mbase + r) * LROW;
            lr[ 0 + lane16] = acc0[r] + b0;
            lr[16 + lane16] = acc1[r] + b1;
            lr[32 + lane16] = acc2[r] + b2;
            lr[48 + lane16] = acc3[r] + b3;
        }
    }
    __syncthreads();

    // ---- per-token softmax / lse / top-K (one token per thread, tid<128) ----
    if (tid < TOK_PER_BLOCK) {
        const int tok = tok0 + tid;
        const float* lr = logits_lds + tid * LROW;
        const float nullv = null_logit[0];

        float lmax = -3.4e38f;
        for (int e = 0; e < E_; ++e) lmax = fmaxf(lmax, lr[e]);

        const float M = fmaxf(lmax, nullv);
        float sumM = 0.0f, sumR = 0.0f;
        for (int e = 0; e < E_; ++e) {
            float le = lr[e];
            sumM += __expf(le - M);       // for lse over all 128 logits
            sumR += __expf(le - lmax);    // softmax denom over real experts
        }
        const float lse = M + __logf(sumM + (float)(E_) * __expf(nullv - M));
        atomicAdd(&blk_misc[0], lse * lse);

        const float invR = 1.0f / sumR;
        for (int e = 0; e < E_; ++e)
            atomicAdd(&blk_P[e], __expf(lr[e] - lmax) * invR);

        // top-K: all 64 null logits are equal; real expert wins ties (lower index).
        unsigned long long chosen = 0ull;
        int   sidx[K_];
        float sexp[K_];
        int m = 0;
        float wsum = 0.0f;
        for (int j = 0; j < K_; ++j) {
            float best = -3.4e38f; int bi = 0;
            for (int e = 0; e < E_; ++e) {
                if (!((chosen >> e) & 1ull)) {
                    float v = lr[e];
                    if (v > best) { best = v; bi = e; }  // strict > => lowest index on ties
                }
            }
            if (best >= nullv) {
                chosen |= 1ull << bi;
                sidx[m] = bi;
                float ex = __expf(best - lmax);
                sexp[m] = ex; wsum += ex; ++m;
            } else break;   // remaining maxima only decrease -> rest are null slots
        }
        for (int j = 0; j < m; ++j) atomicAdd(&blk_cnt[sidx[j]], 1.0f);
        atomicAdd(&blk_misc[1], (float)(K_ - m));

        const float invw = (m > 0) ? (1.0f / wsum) : 0.0f;
        const size_t ob = (size_t)tok * K_;
        for (int j = 0; j < K_; ++j) {
            if (j < m) {
                out_idx[ob + j]    = (float)sidx[j];
                out_w[ob + j]      = sexp[j] * invw;   // renormalized over selected reals
                out_isnull[ob + j] = 0.0f;
            } else {
                out_idx[ob + j]    = (float)(E_ + (j - m));  // null ids 64,65,...
                out_w[ob + j]      = 0.0f;
                out_isnull[ob + j] = 1.0f;
            }
        }
    }

    __syncthreads();
    if (tid < E_) {
        atomicAdd(&g_acc[tid],      blk_P[tid]);
        atomicAdd(&g_acc[E_ + tid], blk_cnt[tid]);
    }
    if (tid == 0) atomicAdd(&g_acc[128], blk_misc[0]);
    if (tid == 1) atomicAdd(&g_acc[129], blk_misc[1]);
}

__global__ void moe_finalize(const float* __restrict__ g, float* __restrict__ aux_out) {
    if (threadIdx.x == 0) {
        float S = 0.0f;
        for (int e = 0; e < E_; ++e) S += g[E_ + e];
        const float Sc = fmaxf(S, 1e-6f);
        float lbal = 0.0f;
        for (int e = 0; e < E_; ++e)
            lbal += (g[E_ + e] / Sc) * (g[e] * (1.0f / (float)NTOK));
        lbal *= (float)E_;
        const float nullrate = g[129] * (1.0f / (float)BTK);
        const float lnull = (nullrate - 0.5f) * (nullrate - 0.5f);
        const float lz = g[128] * (1.0f / (float)NTOK);
        aux_out[0] = 0.02f * lbal + 0.001f * lz + 0.01f * lnull;
    }
}

extern "C" void kernel_launch(void* const* d_in, const int* in_sizes, int n_in,
                              void* d_out, int out_size, void* d_ws, size_t ws_size,
                              hipStream_t stream) {
    const float* x     = (const float*)d_in[0];   // [4,4096,2048]
    const float* gw    = (const float*)d_in[1];   // [64,2048]
    const float* bias  = (const float*)d_in[2];   // [64]
    const float* nullv = (const float*)d_in[3];   // scalar
    float* out = (float*)d_out;                   // idx | w | is_null | aux
    float* g   = (float*)d_ws;                    // 130 floats of accumulators

    moe_zero_acc<<<1, 256, 0, stream>>>(g);
    moe_gate_main<<<NTOK / TOK_PER_BLOCK, 256, 0, stream>>>(
        x, gw, bias, nullv,
        out, out + BTK, out + 2 * BTK, g);
    moe_finalize<<<1, 32, 0, stream>>>(g, out + 3 * BTK);
}
